// SE_Temporal_46789373723271
// MI455X (gfx1250) — matrix operand order repaired and checked by reading, compiled, not run on hardware
//
#include <hip/hip_runtime.h>

// ---------------------------------------------------------------------------
// Types for CDNA5 WMMA (wave32): A/B are 16 bf16 per lane, C/D are 8 f32.
// ---------------------------------------------------------------------------
typedef __attribute__((ext_vector_type(16))) __bf16       v16bf;
typedef __attribute__((ext_vector_type(8)))  float        v8f;
typedef __attribute__((ext_vector_type(8)))  unsigned int v8u;
typedef __attribute__((ext_vector_type(4)))  unsigned int v4u;
typedef __attribute__((ext_vector_type(4)))  unsigned int u32x4;
typedef __attribute__((ext_vector_type(8)))  int          i32x8;
typedef __attribute__((ext_vector_type(4)))  int          i32x4;

union FragA { v16bf bf; struct { v4u lo, hi; } p; };
union FragB { v16bf bf; v8u u; };

__device__ __forceinline__ unsigned short f2bf(float f) {
  unsigned u = __float_as_uint(f);
  return (unsigned short)((u + 0x7FFFu + ((u >> 16) & 1u)) >> 16);  // RNE
}

// Problem constants
#define NN   64
#define CC   64
#define TT   300
#define VV   25
#define TP   306          // T padded by 3 on each side for dilations 1..3
#define MTOT (NN*TT*VV)   // 480000 GEMM rows for the conv
#define BCV_SLAB (6*4*32*16)   // bf16 elements of one dilation's B fragments

// ---------------------------------------------------------------------------
// TDM: async 1-D copy global->LDS (8-byte elements), issued by one wave.
// D# group0: count=1 | lds_addr | global_addr(57b) | type=2.
// D# group1: data_size=8B, tensor_dim0=tile_dim0=nelem, dim1=1, stride=nelem.
// ---------------------------------------------------------------------------
__device__ __forceinline__ void tdm_load_to_lds(unsigned lds_off,
                                                const void* gsrc,
                                                unsigned nbytes) {
  unsigned long long ga = (unsigned long long)(uintptr_t)gsrc;
  unsigned ne = nbytes >> 3;                       // 8-byte elements (< 65536)
  u32x4 g0;
  g0[0] = 0x1u;                                    // count = 1 valid descriptor
  g0[1] = lds_off;                                 // lds_addr
  g0[2] = (unsigned)ga;                            // global_addr[31:0]
  g0[3] = (unsigned)(ga >> 32) | 0x80000000u;      // global_addr[56:32] | type=2
  i32x8 g1;
  g1[0] = (int)(3u << 16);                         // data_size = 3 (8 bytes)
  g1[1] = (int)((ne & 0xFFFFu) << 16);             // tensor_dim0[15:0]
  g1[2] = (int)((ne >> 16) | (1u << 16));          // tensor_dim0[31:16], dim1=1
  g1[3] = (int)((ne & 0xFFFFu) << 16);             // tile_dim0 = ne
  g1[4] = 0;                                       // tile_dim1/2 unused
  g1[5] = (int)ne;                                 // tensor_dim0_stride[31:0]
  g1[6] = 0;
  g1[7] = 0;
  i32x4 z4 = {0, 0, 0, 0};
#if defined(__clang_major__) && __clang_major__ >= 23
  i32x8 z8 = {0, 0, 0, 0, 0, 0, 0, 0};
  __builtin_amdgcn_tensor_load_to_lds(g0, g1, z4, z4, z8, 0);
#else
  __builtin_amdgcn_tensor_load_to_lds(g0, g1, z4, z4, 0);
#endif
}

// ---------------------------------------------------------------------------
// K0: pre-swizzle B fragments into per-lane WMMA layout; zero BN stat buffer.
// ---------------------------------------------------------------------------
__global__ void k0_prep(const float* __restrict__ w_se,
                        const float* __restrict__ w1,
                        const float* __restrict__ w2,
                        const float* __restrict__ w3,
                        unsigned short* __restrict__ Bse,
                        unsigned short* __restrict__ Bcv,
                        float* __restrict__ gstats) {
  int gid = blockIdx.x * blockDim.x + threadIdx.x;
  int stride = gridDim.x * blockDim.x;
  // Bse: [kt(10)][nt(19)][lane(32)][q(16)]
  for (int e = gid; e < 10*19*32*16; e += stride) {
    int q = e & 15; int r = e >> 4;
    int lane = r & 31; r >>= 5;
    int nt = r % 19; int kt = r / 19;
    int k = kt*32 + (lane >> 4)*8 + (q & 7) + (q >> 3)*16;
    int c = nt*16 + (lane & 15);
    float val = (k < TT && c < TT) ? w_se[c*TT + k] : 0.f;   // B[k][c] = w_se[c][k]
    Bse[e] = f2bf(val);
  }
  // Bcv: [d(3)][kt(6)][nt(4)][lane(32)][q(16)], K = tap*64 + i
  for (int e = gid; e < 3*BCV_SLAB; e += stride) {
    int q = e & 15; int r = e >> 4;
    int lane = r & 31; r >>= 5;
    int nt = r & 3;  r >>= 2;
    int kt = r % 6;  int d = r / 6;
    int k = kt*32 + (lane >> 4)*8 + (q & 7) + (q >> 3)*16;
    int o = nt*16 + (lane & 15);
    int tap = k / 64, i = k % 64;
    const float* w = (d == 0) ? w1 : (d == 1) ? w2 : w3;     // (O,C,3,1)
    Bcv[e] = f2bf(w[(o*CC + i)*3 + tap]);
  }
  for (int e = gid; e < 384; e += stride) gstats[e] = 0.f;   // sum[192], sumsq[192]
}

// ---------------------------------------------------------------------------
// K1: per (n,c) row: temporal diff + mean over V -> bf16 A rows (K padded 320).
// ---------------------------------------------------------------------------
__global__ void k1_diff(const float* __restrict__ x, unsigned short* __restrict__ ybf) {
  __shared__ float rs[TT];
  int rowid = blockIdx.x;                       // n*64 + c
  const float* xr = x + rowid * (TT*VV);
  for (int t = threadIdx.x; t < TT; t += 256) {
    float acc = 0.f;
    #pragma unroll
    for (int v = 0; v < VV; ++v) acc += xr[t*VV + v];
    rs[t] = acc;
  }
  __syncthreads();
  unsigned short* yr = ybf + rowid * 320;
  for (int t = threadIdx.x; t < 320; t += 256) {
    float val = (t < TT-1) ? (rs[t+1] - rs[t]) * (1.0f/VV) : 0.f;
    yr[t] = f2bf(val);
  }
}

// ---------------------------------------------------------------------------
// K2: SE GEMM  s = sigmoid(y @ w_se^T).  Grid (32 M-groups x 19 N-tiles);
//     one wave = one 16x16 D tile; all 10 A and B frags preloaded so the 10
//     WMMAs issue back-to-back after a single wait.
// ---------------------------------------------------------------------------
__global__ void k2_se_gemm(const unsigned short* __restrict__ ybf,
                           const unsigned short* __restrict__ Bse,
                           float* __restrict__ s) {
  int tid = threadIdx.x;
  int lane = tid & 31, wv = tid >> 5;
  int mt = blockIdx.x * 8 + wv;                 // 0..255
  int nt = blockIdx.y;                          // 0..18
  int rw = lane & 15, h = lane >> 4;
  const unsigned short* yrow = ybf + (mt*16 + rw) * 320;
  FragA a[10];
  FragB b[10];
  #pragma unroll
  for (int kt = 0; kt < 10; ++kt) {             // A 16x32: lo=K[8h..], hi=K[16+8h..]
    int k0 = kt*32 + h*8;
    a[kt].p.lo = *(const v4u*)(yrow + k0);
    a[kt].p.hi = *(const v4u*)(yrow + k0 + 16);
    b[kt].u    = *(const v8u*)(Bse + ((kt*19 + nt)*32 + lane)*16);
  }
  v8f acc = {0.f,0.f,0.f,0.f,0.f,0.f,0.f,0.f};
  #pragma unroll
  for (int kt = 0; kt < 10; ++kt)
    acc = __builtin_amdgcn_wmma_f32_16x16x32_bf16(false, a[kt].bf, false, b[kt].bf,
                                                  (short)0, acc, false, false);
  int col = nt*16 + rw;                         // D: col = lane&15, row = r + 8h
  if (col < TT) {
    int mbase = mt*16 + 8*h;
    #pragma unroll
    for (int r = 0; r < 8; ++r)
      s[(mbase + r)*TT + col] = 1.0f / (1.0f + __expf(-acc[r]));
  }
}

// ---------------------------------------------------------------------------
// K3: xg[n][t_pad][v][c] = bf16( x[n][c][t][v] * s[n*C+c][t] ), zero pad rows.
// ---------------------------------------------------------------------------
__global__ void k3_gate(const float* __restrict__ x, const float* __restrict__ s,
                        unsigned short* __restrict__ xg) {
  __shared__ float tile[VV*CC];                 // [v][c]
  int n = blockIdx.x / TP;
  int tp = blockIdx.x % TP;
  int t = tp - 3;
  int tid = threadIdx.x;
  if (t >= 0 && t < TT) {
    for (int idx = tid; idx < VV*CC; idx += 256) {
      int c = idx / VV, v = idx % VV;
      float sv = s[(n*CC + c)*TT + t];
      tile[v*CC + c] = x[((n*CC + c)*TT + t)*VV + v] * sv;
    }
  } else {
    for (int idx = tid; idx < VV*CC; idx += 256) tile[idx] = 0.f;
  }
  __syncthreads();
  unsigned short* dst = xg + ((n*TP + tp)*VV)*CC;
  for (int idx = tid; idx < VV*CC; idx += 256) dst[idx] = f2bf(tile[idx]);
}

// ---------------------------------------------------------------------------
// Conv GEMM A-fragment loads (channel-last xg => two contiguous b128 per frag)
// ---------------------------------------------------------------------------
__device__ __forceinline__ void load_a_frags(const unsigned short* __restrict__ xg,
                                             FragA (&a)[6], int n, int t, int v,
                                             int h, int dil) {
  #pragma unroll
  for (int kt = 0; kt < 6; ++kt) {
    int tap = kt >> 1, cb = (kt & 1) << 5;      // K = tap*64 + channel
    int base = (((n*TP + t + (tap-1)*dil + 3))*VV + v)*CC + cb + h*8;
    a[kt].p.lo = *(const v4u*)(xg + base);
    a[kt].p.hi = *(const v4u*)(xg + base + 16);
  }
}

// ---------------------------------------------------------------------------
// K4: conv pass 1 — per-(d,o) sum/sumsq.  B fragments for the current dilation
// are staged into LDS once per block by the Tensor Data Mover (wave 0 issues
// tensor_load_to_lds, waits on TENSORcnt, barrier publishes to all 8 waves).
// ---------------------------------------------------------------------------
__global__ void k4_conv_stats(const unsigned short* __restrict__ xg,
                              const unsigned short* __restrict__ Bcv,
                              float* __restrict__ gsum, float* __restrict__ gsq) {
  __shared__ __align__(16) unsigned short sB[BCV_SLAB];   // 24.6 KB
  __shared__ float ssum[192], ssq[192];
  int tid = threadIdx.x;
  if (tid < 192) { ssum[tid] = 0.f; ssq[tid] = 0.f; }
  int lane = tid & 31, wv = tid >> 5;
  int mt = blockIdx.x * 8 + wv;
  int rw = lane & 15, h = lane >> 4;
  int m = mt*16 + rw;
  int v = m % VV, t = (m/VV) % TT, n = m / (TT*VV);
  for (int d = 0; d < 3; ++d) {
    __syncthreads();                            // prior-d reads done / stats zeroed
    if (wv == 0) {
      tdm_load_to_lds((unsigned)(uintptr_t)sB, Bcv + d*BCV_SLAB, BCV_SLAB*2);
      __builtin_amdgcn_s_wait_tensorcnt(0);
    }
    __syncthreads();
    FragA a[6];
    load_a_frags(xg, a, n, t, v, h, d + 1);
    #pragma unroll
    for (int nt = 0; nt < 4; ++nt) {
      FragB b[6];
      #pragma unroll
      for (int kt = 0; kt < 6; ++kt)
        b[kt].u = *(const v8u*)(sB + ((kt*4 + nt)*32 + lane)*16);
      v8f acc = {0.f,0.f,0.f,0.f,0.f,0.f,0.f,0.f};
      #pragma unroll
      for (int kt = 0; kt < 6; ++kt)
        acc = __builtin_amdgcn_wmma_f32_16x16x32_bf16(false, a[kt].bf, false, b[kt].bf,
                                                      (short)0, acc, false, false);
      float ps = 0.f, pq = 0.f;
      #pragma unroll
      for (int r = 0; r < 8; ++r) { ps += acc[r]; pq += acc[r]*acc[r]; }
      atomicAdd(&ssum[d*64 + nt*16 + rw], ps);
      atomicAdd(&ssq [d*64 + nt*16 + rw], pq);
    }
  }
  __syncthreads();
  if (tid < 192) { atomicAdd(&gsum[tid], ssum[tid]); atomicAdd(&gsq[tid], ssq[tid]); }
}

// K5: per (d,o) BN affine from global stats (conv bias cancels inside BN).
__global__ void k5_bn(const float* __restrict__ gstats,
                      const float* __restrict__ g1, const float* __restrict__ be1,
                      const float* __restrict__ g2, const float* __restrict__ be2,
                      const float* __restrict__ g3, const float* __restrict__ be3,
                      float* __restrict__ scale, float* __restrict__ shift) {
  int tid = threadIdx.x;
  if (tid >= 192) return;
  int d = tid / 64, o = tid % 64;
  const float inv = 1.0f / (float)MTOT;
  float mean = gstats[tid] * inv;
  float var  = gstats[192 + tid] * inv - mean*mean;
  const float* g  = (d == 0) ? g1  : (d == 1) ? g2  : g3;
  const float* be = (d == 0) ? be1 : (d == 1) ? be2 : be3;
  float scl = g[o] * rsqrtf(var + 1e-5f);
  scale[tid] = scl;
  shift[tid] = be[o] - scl * mean;
}

// ---------------------------------------------------------------------------
// K6: conv pass 2 — recompute (xg L2-resident), fuse BN + 3-dilation sum.
// Output stores vectorized: a lane's 8 D-rows are 8 consecutive floats in the
// NCHW output (16B-aligned) => two global_store_b128 per (lane, N-tile).
// ---------------------------------------------------------------------------
__global__ void k6_conv_out(const unsigned short* __restrict__ xg,
                            const unsigned short* __restrict__ Bcv,
                            const float* __restrict__ scale,
                            const float* __restrict__ shift,
                            float* __restrict__ out) {
  __shared__ __align__(16) unsigned short sB[BCV_SLAB];
  int tid = threadIdx.x;
  int lane = tid & 31, wv = tid >> 5;
  int mt = blockIdx.x * 8 + wv;
  int rw = lane & 15, h = lane >> 4;
  int m = mt*16 + rw;
  int v = m % VV, t = (m/VV) % TT, n = m / (TT*VV);
  v8f tot[4];
  #pragma unroll
  for (int nt = 0; nt < 4; ++nt) {
    int o = nt*16 + rw;
    float sh = shift[o] + shift[64 + o] + shift[128 + o];
    #pragma unroll
    for (int r = 0; r < 8; ++r) tot[nt][r] = sh;
  }
  for (int d = 0; d < 3; ++d) {
    __syncthreads();
    if (wv == 0) {
      tdm_load_to_lds((unsigned)(uintptr_t)sB, Bcv + d*BCV_SLAB, BCV_SLAB*2);
      __builtin_amdgcn_s_wait_tensorcnt(0);
    }
    __syncthreads();
    FragA a[6];
    load_a_frags(xg, a, n, t, v, h, d + 1);
    #pragma unroll
    for (int nt = 0; nt < 4; ++nt) {
      FragB b[6];
      #pragma unroll
      for (int kt = 0; kt < 6; ++kt)
        b[kt].u = *(const v8u*)(sB + ((kt*4 + nt)*32 + lane)*16);
      v8f acc = {0.f,0.f,0.f,0.f,0.f,0.f,0.f,0.f};
      #pragma unroll
      for (int kt = 0; kt < 6; ++kt)
        acc = __builtin_amdgcn_wmma_f32_16x16x32_bf16(false, a[kt].bf, false, b[kt].bf,
                                                      (short)0, acc, false, false);
      float scl = scale[d*64 + nt*16 + rw];
      #pragma unroll
      for (int r = 0; r < 8; ++r) tot[nt][r] += scl * acc[r];
    }
  }
  int mrow = mt*16 + 8*h;                       // D rows = r + 8h
  int nA = mrow / (TT*VV), nB = (mrow + 7) / (TT*VV);
  if (nA == nB) {                               // 8-run stays inside one n plane
    int l0 = mrow - nA*(TT*VV);                 // = t*25+v, contiguous in m
    #pragma unroll
    for (int nt = 0; nt < 4; ++nt) {
      int o = nt*16 + rw;
      float* p = out + (nA*CC + o)*(TT*VV) + l0;    // 16B aligned (see analysis)
      float4 lo4; lo4.x = tot[nt][0]; lo4.y = tot[nt][1]; lo4.z = tot[nt][2]; lo4.w = tot[nt][3];
      float4 hi4; hi4.x = tot[nt][4]; hi4.y = tot[nt][5]; hi4.z = tot[nt][6]; hi4.w = tot[nt][7];
      *(float4*)(p)     = lo4;
      *(float4*)(p + 4) = hi4;
    }
  } else {                                      // rare: straddles an n boundary
    #pragma unroll
    for (int nt = 0; nt < 4; ++nt) {
      int o = nt*16 + rw;
      #pragma unroll
      for (int r = 0; r < 8; ++r) {
        int m2 = mrow + r;
        int v2 = m2 % VV, t2 = (m2/VV) % TT, n2 = m2 / (TT*VV);
        out[((n2*CC + o)*TT + t2)*VV + v2] = tot[nt][r];
      }
    }
  }
}

// ---------------------------------------------------------------------------
extern "C" void kernel_launch(void* const* d_in, const int* in_sizes, int n_in,
                              void* d_out, int out_size, void* d_ws, size_t ws_size,
                              hipStream_t stream) {
  const float* x    = (const float*)d_in[0];
  const float* w_se = (const float*)d_in[1];
  const float* w1   = (const float*)d_in[2];
  const float* w2   = (const float*)d_in[4];
  const float* w3   = (const float*)d_in[6];
  const float* g1   = (const float*)d_in[8];
  const float* be1  = (const float*)d_in[9];
  const float* g2   = (const float*)d_in[10];
  const float* be2  = (const float*)d_in[11];
  const float* g3   = (const float*)d_in[12];
  const float* be3  = (const float*)d_in[13];
  float* out = (float*)d_out;

  char* ws = (char*)d_ws;                               // ~70.5 MB used
  unsigned short* ybf = (unsigned short*)(ws + 0);          // 4096x320 bf16
  unsigned short* Bse = (unsigned short*)(ws + 2621440);    // SE B frags
  float*          s   = (float*)(ws + 2816000);             // 4096x300 f32
  unsigned short* xg  = (unsigned short*)(ws + 7731200);    // 64x306x25x64 bf16
  unsigned short* Bcv = (unsigned short*)(ws + 70400000);   // conv B frags
  float*       gstats = (float*)(ws + 70473728);            // sum[192]+sumsq[192]
  float*        scale = (float*)(ws + 70475264);            // 192
  float*        shift = scale + 192;                        // 192

  k0_prep      <<<128, 256, 0, stream>>>(w_se, w1, w2, w3, Bse, Bcv, gstats);
  k1_diff      <<<4096, 256, 0, stream>>>(x, ybf);
  k2_se_gemm   <<<dim3(32, 19), 256, 0, stream>>>(ybf, Bse, s);
  k3_gate      <<<NN*TP, 256, 0, stream>>>(x, s, xg);
  k4_conv_stats<<<MTOT/16/8, 256, 0, stream>>>(xg, Bcv, gstats, gstats + 192);
  k5_bn        <<<1, 192, 0, stream>>>(gstats, g1, be1, g2, be2, g3, be3, scale, shift);
  k6_conv_out  <<<MTOT/16/8, 256, 0, stream>>>(xg, Bcv, scale, shift, out);
}
